// ptm_22239340659340
// MI455X (gfx1250) — matrix operand order repaired and checked by reading, not compiled, not run on hardware
//
#include <hip/hip_runtime.h>
#include <cstdint>
#include <cstddef>

// ---------------------------------------------------------------------------
// Problem constants (from reference): B=128, N=128, T=336, H=64, K=4, F=720
// BN = 16384 tokens. K padded 336->352 (11 x 32), F padded 720->768 for WMMA.
// ---------------------------------------------------------------------------
#define BN    16384
#define T_IN  336
#define TP    352          // padded K for 16x16x32 WMMA
#define NF    720
#define NP    768          // padded N (6 x 128 workgroup tiles)
#define HH    64

typedef __attribute__((ext_vector_type(16))) __bf16        v16bf;
typedef __attribute__((ext_vector_type(8)))  float         v8f;
typedef __attribute__((ext_vector_type(8)))  unsigned short ush8;
typedef __attribute__((ext_vector_type(4)))  int           v4i;

#define AS1 __attribute__((address_space(1)))
#define AS3 __attribute__((address_space(3)))

union FragU { struct { ush8 lo, hi; } s; v16bf v; };

// --- CDNA5 async global->LDS path (ASYNCcnt), with safe fallback -----------
#if defined(__AMDGCN__) && __has_builtin(__builtin_amdgcn_global_load_async_to_lds_b128)
#define HAVE_ASYNC_LDS 1
#else
#define HAVE_ASYNC_LDS 0
#endif

#if HAVE_ASYNC_LDS
__device__ __forceinline__ void async_cp16(const unsigned short* g, unsigned short* l) {
    // 16 bytes, memory -> LDS directly (global_load_async_to_lds_b128)
    __builtin_amdgcn_global_load_async_to_lds_b128(
        (AS1 v4i*)(uintptr_t)g,     // global generic addresses == AS1 addresses
        (AS3 v4i*)l,                // explicit addrspacecast generic -> LDS
        0, 0);
}
__device__ __forceinline__ void wait_async0() {
#if __has_builtin(__builtin_amdgcn_s_wait_asynccnt)
    __builtin_amdgcn_s_wait_asynccnt(0);
#else
    asm volatile("s_wait_asynccnt 0x0" ::: "memory");
#endif
}
#endif

__device__ __forceinline__ unsigned short f2bf(float x) {
    unsigned int u = __float_as_uint(x);
    unsigned int r = u + 0x7FFFu + ((u >> 16) & 1u);   // round-to-nearest-even
    if ((u & 0x7F800000u) == 0x7F800000u) r = u;       // inf/nan passthrough
    return (unsigned short)(r >> 16);
}

__device__ __forceinline__ float redux16(float v) {    // sum over 16-lane group
    v += __shfl_xor(v, 1, 32);
    v += __shfl_xor(v, 2, 32);
    v += __shfl_xor(v, 4, 32);
    v += __shfl_xor(v, 8, 32);
    return v;
}

__device__ __forceinline__ float softplusf(float v) {
    return v > 20.f ? v : log1pf(expf(v));
}

// ---------------------------------------------------------------------------
// Prep: fitter W1 [720,336] -> bf16 [768,352] zero-padded + rowsums S;
//       wg W1 [64,336] -> bf16 [64,352] + rowsums Sw.
// ---------------------------------------------------------------------------
__global__ void prep_w1(const float* __restrict__ f0, const float* __restrict__ f1,
                        const float* __restrict__ f2, const float* __restrict__ f3,
                        const float* __restrict__ wg,
                        unsigned short* __restrict__ w1b,
                        unsigned short* __restrict__ w1bwg,
                        float* __restrict__ S, float* __restrict__ Sw) {
    int idx = blockIdx.x * blockDim.x + threadIdx.x;
    if (idx < 4 * NP) {
        int fit = idx / NP, r = idx % NP;
        const float* src = (fit == 0) ? f0 : (fit == 1) ? f1 : (fit == 2) ? f2 : f3;
        unsigned short* dst = w1b + ((size_t)fit * NP + r) * TP;
        float s = 0.f;
        for (int c = 0; c < TP; ++c) {
            float v = (r < NF && c < T_IN) ? src[(size_t)r * T_IN + c] : 0.f;
            s += v;
            dst[c] = f2bf(v);
        }
        S[idx] = s;
    } else if (idx < 4 * NP + HH) {
        int r = idx - 4 * NP;
        unsigned short* dst = w1bwg + (size_t)r * TP;
        float s = 0.f;
        for (int c = 0; c < TP; ++c) {
            float v = (c < T_IN) ? wg[(size_t)r * T_IN + c] : 0.f;
            s += v;
            dst[c] = f2bf(v);
        }
        Sw[r] = s;
    }
}

// Prep: zero-pad fitter b1 -> [4][768], fitter W2 -> [4][3][768]
__global__ void prep_pad(const float* __restrict__ b1_0, const float* __restrict__ b1_1,
                         const float* __restrict__ b1_2, const float* __restrict__ b1_3,
                         const float* __restrict__ W2_0, const float* __restrict__ W2_1,
                         const float* __restrict__ W2_2, const float* __restrict__ W2_3,
                         float* __restrict__ b1pad, float* __restrict__ w2pad) {
    int idx = blockIdx.x * blockDim.x + threadIdx.x;
    if (idx >= 4 * NP) return;
    int fit = idx / NP, j = idx % NP;
    const float* b1 = (fit == 0) ? b1_0 : (fit == 1) ? b1_1 : (fit == 2) ? b1_2 : b1_3;
    const float* W2 = (fit == 0) ? W2_0 : (fit == 1) ? W2_1 : (fit == 2) ? W2_2 : W2_3;
    int dg = (fit == 1) ? 2 : (fit == 2) ? 3 : 1;
    b1pad[idx] = (j < NF) ? b1[j] : 0.f;
#pragma unroll
    for (int d = 0; d < 3; ++d)
        w2pad[((size_t)fit * 3 + d) * NP + j] =
            (j < NF && d < dg) ? W2[(size_t)d * NF + j] : 0.f;
}

// Prep: eps fp32 [BN,K,336] -> bf16 [BN*K, 352] zero-padded
__global__ void prep_eps(const float* __restrict__ eps, unsigned short* __restrict__ epsb) {
    size_t idx = (size_t)blockIdx.x * blockDim.x + threadIdx.x;
    const size_t total = (size_t)BN * 4 * TP;
    if (idx >= total) return;
    int t = (int)(idx % TP);
    size_t r = idx / TP;
    float v = (t < T_IN) ? eps[r * T_IN + t] : 0.f;
    epsb[idx] = f2bf(v);
}

// ---------------------------------------------------------------------------
// mu/sigma generator MLPs + softplus: one 64-thread block per token.
// ---------------------------------------------------------------------------
__global__ __launch_bounds__(64) void musig_kernel(
    const float* __restrict__ x,
    const float* __restrict__ muW1, const float* __restrict__ mub1,
    const float* __restrict__ muW2, const float* __restrict__ mub2,
    const float* __restrict__ sgW1, const float* __restrict__ sgb1,
    const float* __restrict__ sgW2, const float* __restrict__ sgb2,
    float* __restrict__ spmu, float* __restrict__ spsig) {
    __shared__ float xs[T_IN];
    __shared__ float hmu[HH];
    __shared__ float hsg[HH];
    const int bn = blockIdx.x;
    const int j = threadIdx.x;
    for (int t = j; t < T_IN; t += 64) xs[t] = x[(size_t)bn * T_IN + t];
    __syncthreads();
    float am = mub1[j], asg = sgb1[j];
    const float* wm = muW1 + (size_t)j * T_IN;
    const float* wsp = sgW1 + (size_t)j * T_IN;
    for (int t = 0; t < T_IN; ++t) {
        float xv = xs[t];
        am = fmaf(xv, wm[t], am);
        asg = fmaf(xv, wsp[t], asg);
    }
    hmu[j] = fmaxf(am, 0.f);
    hsg[j] = fmaxf(asg, 0.f);
    __syncthreads();
    if (j < 8) {
        int k = j & 3;
        bool isMu = j < 4;
        const float* W2 = isMu ? muW2 : sgW2;
        const float* b2 = isMu ? mub2 : sgb2;
        const float* h = isMu ? hmu : hsg;
        float a = b2[k];
        for (int q = 0; q < HH; ++q) a = fmaf(h[q], W2[k * HH + q], a);
        (isMu ? spmu : spsig)[bn * 4 + k] = softplusf(a);
    }
}

// ---------------------------------------------------------------------------
// Fitter GEMM + fused second layer.
// C = eps_i @ W1_i^T over 128x128 tiles; epilogue:
//   h = relu(sig_r*C + mu_r*S[col] + b1[col]);  coeff[d] += h * W2[d][col]
// Grid (Mtiles=128, Ntiles=6, fitter=4), block 256 = 8 waves (4M x 2N),
// wave tile 32x64 -> 8 x v_wmma_f32_16x16x32_bf16 per k-step (11 steps).
// A/B tiles staged via global_load_async_to_lds_b128 (ASYNCcnt) when available.
// ---------------------------------------------------------------------------
__global__ __launch_bounds__(256) void gemm_fitter(
    const unsigned short* __restrict__ epsb,
    const unsigned short* __restrict__ w1b,
    const float* __restrict__ Sarr, const float* __restrict__ b1pad,
    const float* __restrict__ w2pad,
    const float* __restrict__ spmu, const float* __restrict__ spsig,
    float* __restrict__ coeffs) {
    const int fit = blockIdx.z;
    const int m0 = blockIdx.x * 128;
    const int n0 = blockIdx.y * 128;
    __shared__ __align__(16) unsigned short As[128 * 40];
    __shared__ __align__(16) unsigned short Bs[128 * 40];
    __shared__ float csum[128 * 3];
    const int tid = threadIdx.x;
    const int lane = tid & 31;
    const int wave = tid >> 5;
    const int wm = wave & 3;       // 4 M-waves
    const int wn = wave >> 2;      // 2 N-waves
    const int l16 = lane & 15;
    const int kg = lane >> 4;      // k-half group per documented bf16 layouts

    for (int i = tid; i < 128 * 3; i += 256) csum[i] = 0.f;

    v8f acc[2][4] = {};

    const int lr = tid >> 1;              // tile row this thread stages
    const int lk = (tid & 1) * 16;        // 16-element half of the 32-wide slab
    const unsigned short* aG = epsb + ((size_t)(m0 + lr) * 4 + fit) * TP + lk;
    const unsigned short* bG = w1b + ((size_t)fit * NP + (n0 + lr)) * TP + lk;
    unsigned short* aL = &As[lr * 40 + lk];
    unsigned short* bL = &Bs[lr * 40 + lk];

    for (int k0 = 0; k0 < TP; k0 += 32) {
#if HAVE_ASYNC_LDS
        __syncthreads();                       // prior LDS readers done
        async_cp16(aG + k0, aL);
        async_cp16(aG + k0 + 8, aL + 8);
        async_cp16(bG + k0, bL);
        async_cp16(bG + k0 + 8, bL + 8);
        if (k0 + 32 < TP) {
            __builtin_prefetch(aG + k0 + 32, 0, 3);   // global_prefetch_b8
            __builtin_prefetch(bG + k0 + 32, 0, 3);
        }
        wait_async0();                         // s_wait_asynccnt 0
        __syncthreads();                       // all waves' tiles resident
#else
        ush8 a0 = *(const ush8*)(aG + k0);
        ush8 a1 = *(const ush8*)(aG + k0 + 8);
        ush8 b0 = *(const ush8*)(bG + k0);
        ush8 b1v = *(const ush8*)(bG + k0 + 8);
        if (k0 + 32 < TP) {
            __builtin_prefetch(aG + k0 + 32, 0, 3);
            __builtin_prefetch(bG + k0 + 32, 0, 3);
        }
        __syncthreads();
        *(ush8*)aL = a0;
        *(ush8*)(aL + 8) = a1;
        *(ush8*)bL = b0;
        *(ush8*)(bL + 8) = b1v;
        __syncthreads();
#endif

        v16bf afrag[2], bfrag[4];
#pragma unroll
        for (int mm = 0; mm < 2; ++mm) {
            int row = wm * 32 + mm * 16 + l16;
            FragU u;                                  // A: contiguous K half per lane group
            u.s.lo = *(const ush8*)&As[row * 40 + kg * 16];
            u.s.hi = *(const ush8*)&As[row * 40 + kg * 16 + 8];
            afrag[mm] = u.v;
        }
#pragma unroll
        for (int nn = 0; nn < 4; ++nn) {
            int col = wn * 64 + nn * 16 + l16;
            FragU u;                                  // B: contiguous K half per lane group
            u.s.lo = *(const ush8*)&Bs[col * 40 + kg * 16];
            u.s.hi = *(const ush8*)&Bs[col * 40 + kg * 16 + 8];
            bfrag[nn] = u.v;
        }
#pragma unroll
        for (int mm = 0; mm < 2; ++mm)
#pragma unroll
            for (int nn = 0; nn < 4; ++nn)
                acc[mm][nn] = __builtin_amdgcn_wmma_f32_16x16x32_bf16(
                    false, afrag[mm], false, bfrag[nn], (short)0, acc[mm][nn],
                    false, false);
    }

    // Fused epilogue: affine + ReLU + dot with padded W2 rows, lane-reduce, LDS acc.
#pragma unroll
    for (int mm = 0; mm < 2; ++mm) {
#pragma unroll
        for (int nn = 0; nn < 4; ++nn) {
            int colL = n0 + wn * 64 + nn * 16 + l16;
            float Sv = Sarr[fit * NP + colL];
            float b1c = b1pad[fit * NP + colL];
            float w20 = w2pad[((size_t)fit * 3 + 0) * NP + colL];
            float w21 = w2pad[((size_t)fit * 3 + 1) * NP + colL];
            float w22 = w2pad[((size_t)fit * 3 + 2) * NP + colL];
#pragma unroll
            for (int g = 0; g < 8; ++g) {
                int rowT = wm * 32 + mm * 16 + kg * 8 + g;
                int grow = m0 + rowT;
                float sig = spsig[grow * 4 + fit];
                float mu = spmu[grow * 4 + fit];
                float cv = acc[mm][nn][g];
                float hp = fmaxf(fmaf(sig, cv, fmaf(mu, Sv, b1c)), 0.f);
                float r0 = redux16(hp * w20);
                float r1 = redux16(hp * w21);
                float r2 = redux16(hp * w22);
                if (l16 == 0) {
                    atomicAdd(&csum[rowT * 3 + 0], r0);
                    atomicAdd(&csum[rowT * 3 + 1], r1);
                    atomicAdd(&csum[rowT * 3 + 2], r2);
                }
            }
        }
    }
    __syncthreads();
    if (tid < 128) {
        size_t grow = (size_t)(m0 + tid);
#pragma unroll
        for (int d = 0; d < 3; ++d)
            atomicAdd(&coeffs[grow * 12 + fit * 3 + d], csum[tid * 3 + d]);
    }
}

// ---------------------------------------------------------------------------
// Weight-generator GEMM (M=65536 z-rows, N=64) + fused logits reduction.
// Block 256 = 8 waves, each wave 16 rows x 64 cols (4 WMMAs per k-step).
// ---------------------------------------------------------------------------
__global__ __launch_bounds__(256) void gemm_wg(
    const unsigned short* __restrict__ epsb,
    const unsigned short* __restrict__ w1bwg,
    const float* __restrict__ Sw, const float* __restrict__ b1w,
    const float* __restrict__ w2w, const float* __restrict__ b2w,
    const float* __restrict__ spmu, const float* __restrict__ spsig,
    float* __restrict__ logits) {
    const int m0 = blockIdx.x * 128;
    __shared__ __align__(16) unsigned short As[128 * 40];
    __shared__ __align__(16) unsigned short Bs[64 * 40];
    __shared__ float lsum[128];
    const int tid = threadIdx.x;
    const int lane = tid & 31;
    const int wave = tid >> 5;
    const int l16 = lane & 15;
    const int kg = lane >> 4;
    if (tid < 128) lsum[tid] = 0.f;

    v8f acc[4] = {};
    const int lr = tid >> 1;
    const int lk = (tid & 1) * 16;
    const unsigned short* aG = epsb + (size_t)(m0 + lr) * TP + lk;
    const unsigned short* bG = w1bwg + (size_t)(lr & 63) * TP + lk;
    unsigned short* aL = &As[lr * 40 + lk];
    unsigned short* bL = &Bs[(lr & 63) * 40 + lk];

    for (int k0 = 0; k0 < TP; k0 += 32) {
#if HAVE_ASYNC_LDS
        __syncthreads();
        async_cp16(aG + k0, aL);
        async_cp16(aG + k0 + 8, aL + 8);
        if (tid < 128) {
            async_cp16(bG + k0, bL);
            async_cp16(bG + k0 + 8, bL + 8);
        }
        if (k0 + 32 < TP) __builtin_prefetch(aG + k0 + 32, 0, 3);
        wait_async0();
        __syncthreads();
#else
        ush8 a0 = *(const ush8*)(aG + k0);
        ush8 a1 = *(const ush8*)(aG + k0 + 8);
        ush8 b0, b1v;
        if (tid < 128) {
            b0 = *(const ush8*)(bG + k0);
            b1v = *(const ush8*)(bG + k0 + 8);
        }
        if (k0 + 32 < TP) __builtin_prefetch(aG + k0 + 32, 0, 3);
        __syncthreads();
        *(ush8*)aL = a0;
        *(ush8*)(aL + 8) = a1;
        if (tid < 128) {
            *(ush8*)bL = b0;
            *(ush8*)(bL + 8) = b1v;
        }
        __syncthreads();
#endif

        FragU ua;
        int row = wave * 16 + l16;
        ua.s.lo = *(const ush8*)&As[row * 40 + kg * 16];
        ua.s.hi = *(const ush8*)&As[row * 40 + kg * 16 + 8];
        v16bf af = ua.v;
#pragma unroll
        for (int nn = 0; nn < 4; ++nn) {
            FragU ub;
            int col = nn * 16 + l16;
            ub.s.lo = *(const ush8*)&Bs[col * 40 + kg * 16];
            ub.s.hi = *(const ush8*)&Bs[col * 40 + kg * 16 + 8];
            acc[nn] = __builtin_amdgcn_wmma_f32_16x16x32_bf16(
                false, af, false, ub.v, (short)0, acc[nn], false, false);
        }
    }

#pragma unroll
    for (int nn = 0; nn < 4; ++nn) {
        int col = nn * 16 + l16;
        float Sv = Sw[col], b1c = b1w[col], w2v = w2w[col];
#pragma unroll
        for (int g = 0; g < 8; ++g) {
            int rowT = wave * 16 + kg * 8 + g;
            int grow = m0 + rowT;                 // z-row r = bn*4+k; sp* flat [BN][4]
            float sig = spsig[grow];
            float mu = spmu[grow];
            float hp = fmaxf(fmaf(sig, acc[nn][g], fmaf(mu, Sv, b1c)), 0.f);
            float r = redux16(hp * w2v);
            if (l16 == 0) atomicAdd(&lsum[rowT], r);
        }
    }
    __syncthreads();
    if (tid < 128) logits[m0 + tid] = lsum[tid] + b2w[0];
}

// Softmax over K=4 mixture dim
__global__ void softmax_kernel(const float* __restrict__ logits, float* __restrict__ wmix) {
    int bn = blockIdx.x * blockDim.x + threadIdx.x;
    if (bn >= BN) return;
    float l0 = logits[bn * 4 + 0], l1 = logits[bn * 4 + 1];
    float l2 = logits[bn * 4 + 2], l3 = logits[bn * 4 + 3];
    float m = fmaxf(fmaxf(l0, l1), fmaxf(l2, l3));
    float e0 = expf(l0 - m), e1 = expf(l1 - m), e2 = expf(l2 - m), e3 = expf(l3 - m);
    float inv = 1.f / (e0 + e1 + e2 + e3);
    wmix[bn * 4 + 0] = e0 * inv;
    wmix[bn * 4 + 1] = e1 * inv;
    wmix[bn * 4 + 2] = e2 * inv;
    wmix[bn * 4 + 3] = e3 * inv;
}

// Polynomial evaluation + mixture combine -> out[B,F,N]
__global__ void final_kernel(const float* __restrict__ coeffs, const float* __restrict__ wmix,
                             const float* __restrict__ b2_0, const float* __restrict__ b2_1,
                             const float* __restrict__ b2_2, const float* __restrict__ b2_3,
                             float* __restrict__ out) {
    size_t idx = (size_t)blockIdx.x * blockDim.x + threadIdx.x;
    const size_t total = (size_t)128 * NF * 128;
    if (idx >= total) return;
    int n = (int)(idx & 127);
    size_t rem = idx >> 7;
    int f = (int)(rem % NF);
    int b = (int)(rem / NF);
    int bn = b * 128 + n;
    const float* c = coeffs + (size_t)bn * 12;
    const float* w = wmix + (size_t)bn * 4;
    float t = (float)f * (1.0f / 719.0f);
    float t2 = t * t, t3 = t2 * t;
    float p0 = (c[0] + b2_0[0]) + 0.5f * t;                                   // deg 1, +0.5
    float p1 = fmaf(c[4] + b2_1[1], t, c[3] + b2_1[0]) + 0.5f * t2;           // deg 2, +0.5
    float p2 = fmaf(c[8] + b2_2[2], t2, fmaf(c[7] + b2_2[1], t, c[6] + b2_2[0])) + 0.5f * t3;
    float p3 = (c[9] + b2_3[0]) - 0.5f * t;                                   // deg 1, -0.5
    out[idx] = w[0] * p0 + w[1] * p1 + w[2] * p2 + w[3] * p3;
}

// ---------------------------------------------------------------------------
extern "C" void kernel_launch(void* const* d_in, const int* in_sizes, int n_in,
                              void* d_out, int out_size, void* d_ws, size_t ws_size,
                              hipStream_t stream) {
    (void)in_sizes; (void)n_in; (void)out_size; (void)ws_size;
    const float* x    = (const float*)d_in[0];
    const float* eps  = (const float*)d_in[1];
    const float* muW1 = (const float*)d_in[2];
    const float* mub1 = (const float*)d_in[3];
    const float* muW2 = (const float*)d_in[4];
    const float* mub2 = (const float*)d_in[5];
    const float* sgW1 = (const float*)d_in[6];
    const float* sgb1 = (const float*)d_in[7];
    const float* sgW2 = (const float*)d_in[8];
    const float* sgb2 = (const float*)d_in[9];
    const float* wgW1 = (const float*)d_in[10];
    const float* wgb1 = (const float*)d_in[11];
    const float* wgW2 = (const float*)d_in[12];
    const float* wgb2 = (const float*)d_in[13];
    const float* fW1[4], *fb1[4], *fW2[4], *fb2[4];
    for (int i = 0; i < 4; ++i) {
        fW1[i] = (const float*)d_in[14 + 4 * i];
        fb1[i] = (const float*)d_in[15 + 4 * i];
        fW2[i] = (const float*)d_in[16 + 4 * i];
        fb2[i] = (const float*)d_in[17 + 4 * i];
    }
    float* out = (float*)d_out;

    // Workspace carve-up (~50.2 MB total), 256B aligned.
    char* base = (char*)d_ws;
    size_t off = 0;
    auto carve = [&](size_t bytes) {
        void* p = base + off;
        off = (off + bytes + 255) & ~(size_t)255;
        return p;
    };
    unsigned short* epsb  = (unsigned short*)carve((size_t)BN * 4 * TP * 2);
    unsigned short* w1b   = (unsigned short*)carve((size_t)4 * NP * TP * 2);
    unsigned short* w1bwg = (unsigned short*)carve((size_t)HH * TP * 2);
    float* S      = (float*)carve((size_t)4 * NP * 4);
    float* Sw     = (float*)carve((size_t)HH * 4);
    float* b1pad  = (float*)carve((size_t)4 * NP * 4);
    float* w2pad  = (float*)carve((size_t)4 * 3 * NP * 4);
    float* spmu   = (float*)carve((size_t)BN * 4 * 4);
    float* spsig  = (float*)carve((size_t)BN * 4 * 4);
    float* logits = (float*)carve((size_t)BN * 4 * 4);
    float* wmix   = (float*)carve((size_t)BN * 4 * 4);
    float* coeffs = (float*)carve((size_t)BN * 12 * 4);

    (void)hipMemsetAsync(coeffs, 0, (size_t)BN * 12 * 4, stream);

    prep_w1<<<13, 256, 0, stream>>>(fW1[0], fW1[1], fW1[2], fW1[3], wgW1,
                                    w1b, w1bwg, S, Sw);
    prep_pad<<<12, 256, 0, stream>>>(fb1[0], fb1[1], fb1[2], fb1[3],
                                     fW2[0], fW2[1], fW2[2], fW2[3], b1pad, w2pad);
    {
        size_t total = (size_t)BN * 4 * TP;
        int blocks = (int)((total + 255) / 256);
        prep_eps<<<blocks, 256, 0, stream>>>(eps, epsb);
    }
    musig_kernel<<<BN, 64, 0, stream>>>(x, muW1, mub1, muW2, mub2,
                                        sgW1, sgb1, sgW2, sgb2, spmu, spsig);
    gemm_wg<<<(BN * 4) / 128, 256, 0, stream>>>(epsb, w1bwg, Sw, wgb1, wgW2, wgb2,
                                                spmu, spsig, logits);
    softmax_kernel<<<BN / 256, 256, 0, stream>>>(logits, wmix);
    gemm_fitter<<<dim3(BN / 128, NP / 128, 4), 256, 0, stream>>>(
        epsb, w1b, S, b1pad, w2pad, spmu, spsig, coeffs);
    {
        size_t total = (size_t)128 * NF * 128;
        int blocks = (int)((total + 255) / 256);
        final_kernel<<<blocks, 256, 0, stream>>>(coeffs, wmix,
                                                 fb2[0], fb2[1], fb2[2], fb2[3], out);
    }
}
